// MessagePassingLayer_35081292873864
// MI455X (gfx1250) — matrix operand order repaired and checked by reading, not compiled, not run on hardware
//
#include <hip/hip_runtime.h>
#include <hip/hip_bf16.h>

// ---------------- problem constants (from reference) ----------------
#define HID   128            // H
#define HC    256            // 2*H
#define NN    81             // nodes
#define BB    256            // batch
#define EE    1620           // edges
#define ETIL  102            // ceil(EE/16)
#define NTIL  6              // ceil(NN/16)
#define WPB   4              // waves per block
#define TPW   8              // tiles per wave (edge kernel)
#define NWAVES ((BB * ETIL) / TPW)   // 3264 waves -> 816 blocks
// edge-kernel LDS layout (single segment, base offset 0)
#define W1_OFF 0                      // 65536 B : Wm1^T f16 [128][256]
#define W2_OFF 65536                  // 32768 B : Wm2^T f16 [128][128]
#define A_OFF  98304                  // WPB * 8192 : gathered comb rows f16
#define H_OFF  131072                 // WPB * 4096 : relu hidden f16
#define SMEM_TOT 147456               // 144 KB of the 320 KB WGP pool

typedef _Float16 half_t;
typedef __attribute__((ext_vector_type(16))) _Float16 v16h;
typedef __attribute__((ext_vector_type(8)))  _Float16 v8h;
typedef __attribute__((ext_vector_type(4)))  _Float16 v4h;
typedef __attribute__((ext_vector_type(8)))  float    v8f;

union HFrag { v16h v; v8h h[2]; };

__device__ __forceinline__ v8f wmma16(v16h a, v16h b, v8f c) {
  return __builtin_amdgcn_wmma_f32_16x16x32_f16(
      false, a, false, b, (short)0, c, false, false);
}

// -------- per-node in-degree reciprocal --------
__global__ void counts_kernel(const int* __restrict__ ei, float* __restrict__ inv_cnt) {
  __shared__ float c[128];
  int t = threadIdx.x;
  c[t] = 0.f;
  __syncthreads();
  for (int e = t; e < EE; e += blockDim.x) atomicAdd(&c[ei[EE + e]], 1.f);
  __syncthreads();
  if (t < NN) inv_cnt[t] = 1.f / fmaxf(c[t], 1.f);
}

// -------- weight (K x 128 f32) -> transposed f16 Wt[n][k] --------
__global__ void wconv_kernel(const float* __restrict__ W, half_t* __restrict__ Wt, int K) {
  int idx = blockIdx.x * blockDim.x + threadIdx.x;
  if (idx < K * HID) {
    int k = idx / HID, n = idx % HID;
    Wt[(size_t)n * K + k] = (half_t)W[(size_t)k * HID + n];
  }
}

// -------- node features f32 -> f16 copy --------
__global__ void nfconv_kernel(const float* __restrict__ nf, half_t* __restrict__ nf16) {
  int idx = blockIdx.x * blockDim.x + threadIdx.x;
  if (idx < BB * NN * HID) nf16[idx] = (half_t)nf[idx];
}

// ---- async gather of one 16-row comb tile (16 x 256 f16 = 8 KB) into LDS ----
// uses GLOBAL_LOAD_ASYNC_TO_LDS_B128 (ASYNCcnt); 512 x 16B chunks, 16 per lane
__device__ __forceinline__ void stage_tile_async(const half_t* nf16, const int* ei,
                                                 int tile, char* Abase, int lane) {
  const int b  = tile / ETIL;
  const int et = tile % ETIL;
  const unsigned long long gbase = (unsigned long long)(uintptr_t)nf16;
#pragma unroll 4
  for (int i = 0; i < 16; ++i) {
    int g = i * 32 + lane;
    int m = g >> 5;                 // row in tile
    int c = g & 31;                 // 16B chunk within row (32 per 512B row)
    int e = et * 16 + m; if (e >= EE) e = EE - 1;
    int node = (c < 16) ? ei[e] : ei[EE + e];
    unsigned long long ga = gbase + ((unsigned long long)(unsigned)(b * NN + node) << 8)
                                  + (unsigned)((c & 15) << 4);
    unsigned la = (unsigned)(uintptr_t)(Abase + m * 512 + c * 16); // low 32b = LDS addr
    asm volatile("global_load_async_to_lds_b128 %0, %1, off"
                 :: "v"(la), "v"(ga) : "memory");
  }
}

// ============================ edge message MLP ============================
__global__ __launch_bounds__(128)
void edge_kernel(const half_t* __restrict__ nf16, const int* __restrict__ ei,
                 const half_t* __restrict__ Wt /* Wm1^T || Wm2^T contiguous */,
                 const float* __restrict__ bm1, const float* __restrict__ bm2,
                 float* __restrict__ agg)
{
  __shared__ __align__(16) char smem[SMEM_TOT];
  const int wave = threadIdx.x >> 5;
  const int lane = threadIdx.x & 31;
  const int w    = blockIdx.x * WPB + wave;

  // ---- cooperative: stage Wm1^T||Wm2^T (96 KB) into LDS once per block ----
  {
    const uint4* src = (const uint4*)Wt;
    uint4* dst = (uint4*)smem;
    for (int i = threadIdx.x; i < (98304 / 16); i += 128) dst[i] = src[i];
  }

  char* Abase = smem + A_OFF + wave * 8192;
  const half_t* A  = (const half_t*)Abase;
  half_t* Hh = (half_t*)(smem + H_OFF + wave * 4096);
  const half_t* W1 = (const half_t*)(smem + W1_OFF);
  const half_t* W2 = (const half_t*)(smem + W2_OFF);

  // prologue: kick off gather for this wave's first tile
  stage_tile_async(nf16, ei, w, Abase, lane);
  __syncthreads();                         // weights visible to all waves

  const int mlo  = lane & 15;
  const int hsel = lane >> 4;

  int tile = w;
  for (int it = 0; it < TPW; ++it, tile += NWAVES) {
    const int b  = tile / ETIL;
    const int et = tile % ETIL;

    asm volatile("s_wait_asynccnt 0x0" ::: "memory");  // gather for this tile done

    // ---- matmul1: hidden(16x128) = comb(16x256) @ Wm1 (weights from LDS) ----
    v8f acc[8] = {};
    for (int k0 = 0; k0 < HC; k0 += 32) {
      HFrag a;
      a.h[0] = *(const v8h*)&A[mlo * HC + k0 + hsel * 8];
      a.h[1] = *(const v8h*)&A[mlo * HC + k0 + 16 + hsel * 8];
#pragma unroll
      for (int j = 0; j < 8; ++j) {
        HFrag bf;
        const half_t* wp = W1 + (j * 16 + mlo) * HC + k0 + hsel * 8;
        bf.h[0] = *(const v8h*)wp;
        bf.h[1] = *(const v8h*)(wp + 16);
        acc[j] = wmma16(a.v, bf.v, acc[j]);
      }
    }

    // ---- bias + relu -> LDS (A-layout source for matmul2) ----
#pragma unroll
    for (int j = 0; j < 8; ++j) {
      float bias = bm1[j * 16 + mlo];
#pragma unroll
      for (int r = 0; r < 8; ++r) {
        float v = fmaxf(acc[j][r] + bias, 0.f);
        Hh[(r + 8 * hsel) * HID + j * 16 + mlo] = (half_t)v;
      }
    }
    // all wave DS ops (A reads + Hh stores) complete -> A region reusable,
    // Hh visible across lanes of this wave
    asm volatile("s_wait_dscnt 0x0" ::: "memory");

    // overlap: gather next tile while matmul2 + scatter run
    if (it + 1 < TPW) stage_tile_async(nf16, ei, tile + NWAVES, Abase, lane);

    // ---- matmul2: msgs(16x128) = hidden(16x128) @ Wm2 ----
    v8f acc2[8] = {};
    for (int k0 = 0; k0 < HID; k0 += 32) {
      HFrag a;
      a.h[0] = *(const v8h*)&Hh[mlo * HID + k0 + hsel * 8];
      a.h[1] = *(const v8h*)&Hh[mlo * HID + k0 + 16 + hsel * 8];
#pragma unroll
      for (int j = 0; j < 8; ++j) {
        HFrag bf;
        const half_t* wp = W2 + (j * 16 + mlo) * HID + k0 + hsel * 8;
        bf.h[0] = *(const v8h*)wp;
        bf.h[1] = *(const v8h*)(wp + 16);
        acc2[j] = wmma16(a.v, bf.v, acc2[j]);
      }
    }

    // ---- scatter-add msgs into agg[b, tgt, :] ----
    int  tg[8];
    bool val[8];
#pragma unroll
    for (int r = 0; r < 8; ++r) {
      int e = et * 16 + r + 8 * hsel;
      val[r] = (e < EE);
      tg[r]  = val[r] ? ei[EE + e] : 0;
    }
#pragma unroll
    for (int j = 0; j < 8; ++j) {
      float bias = bm2[j * 16 + mlo];
#pragma unroll
      for (int r = 0; r < 8; ++r) {
        if (val[r])
          atomicAdd(&agg[((size_t)b * NN + tg[r]) * HID + j * 16 + mlo],
                    acc2[j][r] + bias);
      }
    }
  }
}

// ============================ node update + layernorm ============================
__global__ __launch_bounds__(128)
void node_kernel(const float* __restrict__ nf,
                 const half_t* __restrict__ Wu1t, const float* __restrict__ bu1,
                 const half_t* __restrict__ Wu2t, const float* __restrict__ bu2,
                 const float* __restrict__ agg, const float* __restrict__ inv_cnt,
                 const float* __restrict__ gamma, const float* __restrict__ beta,
                 float* __restrict__ out)
{
  __shared__ half_t ldsA[WPB][16][HC];   // staged comb2; reused as f32 x-tile
  __shared__ half_t ldsH[WPB][16][HID];
  __shared__ float  ldsStat[WPB][16][2];

  const int wave = threadIdx.x >> 5;
  const int lane = threadIdx.x & 31;
  const int tile = blockIdx.x * WPB + wave;
  const int b  = tile / NTIL;
  const int nt = tile % NTIL;

  half_t (*A)[HC]   = ldsA[wave];
  half_t (*Hh)[HID] = ldsH[wave];

  for (int q = lane; q < 16 * 64; q += 32) {
    int m  = q >> 6;
    int c4 = q & 63;
    int n  = nt * 16 + m; if (n >= NN) n = NN - 1;
    float4 s;
    if (c4 < 32) {
      s = ((const float4*)(nf + ((size_t)b * NN + n) * HID))[c4];
    } else {
      s = ((const float4*)(agg + ((size_t)b * NN + n) * HID))[c4 & 31];
      float iv = inv_cnt[n];
      s.x *= iv; s.y *= iv; s.z *= iv; s.w *= iv;
    }
    v4h hv = { (half_t)s.x, (half_t)s.y, (half_t)s.z, (half_t)s.w };
    *(v4h*)&A[m][c4 * 4] = hv;
  }
  __syncthreads();

  const int mlo  = lane & 15;
  const int hsel = lane >> 4;

  v8f acc[8] = {};
  for (int k0 = 0; k0 < HC; k0 += 32) {
    HFrag a;
    a.h[0] = *(const v8h*)&A[mlo][k0 + hsel * 8];
    a.h[1] = *(const v8h*)&A[mlo][k0 + 16 + hsel * 8];
#pragma unroll
    for (int j = 0; j < 8; ++j) {
      HFrag bf;
      const half_t* wp = Wu1t + (size_t)(j * 16 + mlo) * HC + k0 + hsel * 8;
      bf.h[0] = *(const v8h*)wp;
      bf.h[1] = *(const v8h*)(wp + 16);
      acc[j] = wmma16(a.v, bf.v, acc[j]);
    }
  }
#pragma unroll
  for (int j = 0; j < 8; ++j) {
    float bias = bu1[j * 16 + mlo];
#pragma unroll
    for (int r = 0; r < 8; ++r) {
      float v = fmaxf(acc[j][r] + bias, 0.f);
      Hh[r + 8 * hsel][j * 16 + mlo] = (half_t)v;
    }
  }
  __syncthreads();

  v8f acc2[8] = {};
  for (int k0 = 0; k0 < HID; k0 += 32) {
    HFrag a;
    a.h[0] = *(const v8h*)&Hh[mlo][k0 + hsel * 8];
    a.h[1] = *(const v8h*)&Hh[mlo][k0 + 16 + hsel * 8];
#pragma unroll
    for (int j = 0; j < 8; ++j) {
      HFrag bf;
      const half_t* wp = Wu2t + (size_t)(j * 16 + mlo) * HID + k0 + hsel * 8;
      bf.h[0] = *(const v8h*)wp;
      bf.h[1] = *(const v8h*)(wp + 16);
      acc2[j] = wmma16(a.v, bf.v, acc2[j]);
    }
  }
  __syncthreads();   // A-stage region free -> reuse as f32 x-tile

  float (*X)[HID] = (float (*)[HID])ldsA[wave];
#pragma unroll
  for (int j = 0; j < 8; ++j) {
    float bias = bu2[j * 16 + mlo];
    int col = j * 16 + mlo;
#pragma unroll
    for (int r = 0; r < 8; ++r) {
      int m = r + 8 * hsel;
      int n = nt * 16 + m; if (n >= NN) n = NN - 1;
      X[m][col] = acc2[j][r] + bias + nf[((size_t)b * NN + n) * HID + col];
    }
  }
  __syncthreads();

  if (lane < 16) {
    float s = 0.f, s2 = 0.f;
    for (int c = 0; c < HID; ++c) { float v = X[lane][c]; s += v; s2 += v * v; }
    float mu  = s * (1.f / HID);
    float var = s2 * (1.f / HID) - mu * mu;
    ldsStat[wave][lane][0] = mu;
    ldsStat[wave][lane][1] = rsqrtf(var + 1e-5f);
  }
  __syncthreads();

  for (int q = lane; q < 16 * 32; q += 32) {
    int m  = q >> 5;
    int c4 = q & 31;
    int n  = nt * 16 + m;
    if (n < NN) {
      float mu = ldsStat[wave][m][0];
      float rs = ldsStat[wave][m][1];
      int c = c4 * 4;
      float4 o;
      o.x = (X[m][c + 0] - mu) * rs * gamma[c + 0] + beta[c + 0];
      o.y = (X[m][c + 1] - mu) * rs * gamma[c + 1] + beta[c + 1];
      o.z = (X[m][c + 2] - mu) * rs * gamma[c + 2] + beta[c + 2];
      o.w = (X[m][c + 3] - mu) * rs * gamma[c + 3] + beta[c + 3];
      ((float4*)(out + ((size_t)b * NN + n) * HID))[c4] = o;
    }
  }
}

// ================================ launcher ================================
extern "C" void kernel_launch(void* const* d_in, const int* in_sizes, int n_in,
                              void* d_out, int out_size, void* d_ws, size_t ws_size,
                              hipStream_t stream) {
  const float* nf   = (const float*)d_in[0];
  const int*   ei   = (const int*)  d_in[1];
  const float* Wm1  = (const float*)d_in[2];
  const float* bm1  = (const float*)d_in[3];
  const float* Wm2  = (const float*)d_in[4];
  const float* bm2  = (const float*)d_in[5];
  const float* Wu1  = (const float*)d_in[6];
  const float* bu1  = (const float*)d_in[7];
  const float* Wu2  = (const float*)d_in[8];
  const float* bu2  = (const float*)d_in[9];
  const float* gamma= (const float*)d_in[10];
  const float* beta = (const float*)d_in[11];
  float* out = (float*)d_out;

  // workspace layout (all offsets 256B-aligned)
  char* ws = (char*)d_ws;
  const size_t aggBytes = (size_t)BB * NN * HID * sizeof(float); // 10,616,832
  float*  agg     = (float*)ws;
  float*  inv_cnt = (float*)(ws + aggBytes);
  half_t* Wm1t    = (half_t*)(ws + aggBytes + 512);              // 64 KB
  half_t* Wm2t    = (half_t*)((char*)Wm1t + (size_t)HC  * HID * 2); // contiguous!
  half_t* Wu1t    = (half_t*)((char*)Wm2t + (size_t)HID * HID * 2);
  half_t* Wu2t    = (half_t*)((char*)Wu1t + (size_t)HC  * HID * 2);
  half_t* nf16    = (half_t*)((char*)Wu2t + (size_t)HID * HID * 2);

  hipMemsetAsync(agg, 0, aggBytes, stream);

  counts_kernel<<<1, 128, 0, stream>>>(ei, inv_cnt);

  wconv_kernel<<<(HC  * HID + 255) / 256, 256, 0, stream>>>(Wm1, Wm1t, HC);
  wconv_kernel<<<(HID * HID + 255) / 256, 256, 0, stream>>>(Wm2, Wm2t, HID);
  wconv_kernel<<<(HC  * HID + 255) / 256, 256, 0, stream>>>(Wu1, Wu1t, HC);
  wconv_kernel<<<(HID * HID + 255) / 256, 256, 0, stream>>>(Wu2, Wu2t, HID);

  nfconv_kernel<<<(BB * NN * HID + 255) / 256, 256, 0, stream>>>(nf, nf16);

  // edge MLP: grid-stride, 8 tiles per wave, weights LDS-resident
  edge_kernel<<<NWAVES / WPB, 32 * WPB, 0, stream>>>(
      nf16, ei, Wm1t, bm1, bm2, agg);

  node_kernel<<<(BB * NTIL) / WPB, 32 * WPB, 0, stream>>>(
      nf, Wu1t, bu1, Wu2t, bu2, agg, inv_cnt, gamma, beta, out);
}
